// VectorQuantizer_84078279787176
// MI455X (gfx1250) — matrix operand order repaired and checked by reading, not compiled, not run on hardware
//
#include <hip/hip_runtime.h>
#include <hip/hip_bf16.h>
#include <math.h>
#include <stdint.h>

// ---------------------------------------------------------------------------
// VQ forward for MI455X (gfx1250): bf16 WMMA GEMMs + fused rmsnorm/argmax
// + async global->LDS staging (ASYNCcnt) with LDS double buffering
// ---------------------------------------------------------------------------

typedef __bf16 bf16_t;
typedef __attribute__((ext_vector_type(16))) __bf16 v16bf;
typedef __attribute__((ext_vector_type(8)))  __bf16 v8bf;
typedef __attribute__((ext_vector_type(8)))  float   v8f;
typedef __attribute__((ext_vector_type(4)))  int     v4i;

constexpr int kB = 8, kD = 512, kL = 4096, kH = 256, kN = 2048;
constexpr int kM = kB * kL;                // 32768 tokens
constexpr float kEps = 1.1920929e-7f;      // float32 eps (matches jnp.finfo)

// workspace layout (bytes)
constexpr size_t WS_WIN  = 0;                                   // bf16 [H][D]
constexpr size_t WS_WOUT = WS_WIN  + (size_t)kH * kD * 2;       // bf16 [D][H]
constexpr size_t WS_CB   = WS_WOUT + (size_t)kD * kH * 2;       // bf16 [N][H]
constexpr size_t WS_H    = WS_CB   + (size_t)kN * kH * 2;       // bf16 [M][H]
constexpr size_t WS_IND  = WS_H    + (size_t)kM * kH * 2;       // int  [M]
constexpr size_t WS_LOSS = WS_IND  + (size_t)kM * 4;            // float[1]

// output layout (floats): out[B*D*L], inds[M] (as float), loss[1]
constexpr size_t OUT_IND  = (size_t)kB * kD * kL;
constexpr size_t OUT_LOSS = OUT_IND + (size_t)kM;

// ------------------------------ helpers ------------------------------------

__device__ inline v8f vzero8() {
  v8f r;
#pragma unroll
  for (int i = 0; i < 8; ++i) r[i] = 0.0f;
  return r;
}

__device__ inline v8f wmma_bf16(v16bf a, v16bf b, v8f c) {
  // D = A(16x32 bf16) * B(32x16 bf16) + C(16x16 f32)
  return __builtin_amdgcn_wmma_f32_16x16x32_bf16(
      /*neg_a=*/false, a, /*neg_b=*/false, b,
      /*c_mod=*/(short)0, c, /*reuse_a=*/false, /*reuse_b=*/false);
}

// A-fragment: lane row = lane%16,
// elements 0..15 = K[kb..kb+15] (contiguous, matches B)
__device__ inline v16bf ld_split(const bf16_t* p) {
  v8bf lo = *(const v8bf*)p;
  v8bf hi = *(const v8bf*)(p + 8);
  v16bf r;
#pragma unroll
  for (int i = 0; i < 8; ++i) { r[i] = lo[i]; r[i + 8] = hi[i]; }
  return r;
}

// B-fragment: lane col = lane%16, 16 contiguous K values per half-wave
__device__ inline v16bf ld_cont(const bf16_t* p) {
  v8bf lo = *(const v8bf*)p;
  v8bf hi = *(const v8bf*)(p + 8);
  v16bf r;
#pragma unroll
  for (int i = 0; i < 8; ++i) { r[i] = lo[i]; r[i + 8] = hi[i]; }
  return r;
}

// --- async global->LDS staging (GLOBAL_LOAD_ASYNC_TO_LDS_B128, ASYNCcnt) ---
#if __has_builtin(__builtin_amdgcn_global_load_async_to_lds_b128) && \
    __has_builtin(__builtin_amdgcn_s_wait_asynccnt)
#define USE_ASYNC_LDS 1
#endif

#ifdef USE_ASYNC_LDS
// Builtin expects v4i* params (per hipcc diagnostic). Produce AS-qualified
// pointers via integer casts: specific->generic is implicitly convertible on
// AMDGPU, so these arguments match either AS-qualified or generic params.
typedef __attribute__((address_space(1))) v4i gbl_v4i;
typedef __attribute__((address_space(3))) v4i lds_v4i;
__device__ inline void lds_copy16(bf16_t* dst_lds, const bf16_t* src_gbl) {
  // low 32 bits of a flat LDS address are the LDS byte offset (ISA 10.2)
  __builtin_amdgcn_global_load_async_to_lds_b128(
      (gbl_v4i*)(uintptr_t)src_gbl,
      (lds_v4i*)(unsigned int)(uintptr_t)dst_lds,
      /*offset=*/0, /*cpol=*/0);
}
template <int N> __device__ inline void lds_copy_wait() {
  __builtin_amdgcn_s_wait_asynccnt(N);
}
#else
__device__ inline void lds_copy16(bf16_t* dst_lds, const bf16_t* src_gbl) {
  *(v8bf*)dst_lds = *(const v8bf*)src_gbl;
}
template <int N> __device__ inline void lds_copy_wait() {}
#endif

// ------------------------------ kernels ------------------------------------

__global__ void k_zero(float* loss) {
  if (threadIdx.x == 0 && blockIdx.x == 0) *loss = 0.0f;
}

// Normalize weights/codebook rows -> bf16. blocks: [0,H) w_in, [H,H+D) w_out,
// [H+D, H+D+N) codebook.
__global__ __launch_bounds__(128) void k_prep(const float* __restrict__ w_in,
                                              const float* __restrict__ cbf,
                                              const float* __restrict__ w_out,
                                              bf16_t* __restrict__ win_h,
                                              bf16_t* __restrict__ wout_h,
                                              bf16_t* __restrict__ cb_h) {
  __shared__ float red[4];
  const int tid = threadIdx.x, lane = tid & 31, wid = tid >> 5;
  const int rowid = blockIdx.x;
  const float* src; bf16_t* dst; int len; int mode;
  if (rowid < kH) {
    src = w_in + (size_t)rowid * kD; dst = win_h + (size_t)rowid * kD;
    len = kD; mode = 0;
  } else if (rowid < kH + kD) {
    const int r = rowid - kH;
    src = w_out + (size_t)r * kH; dst = wout_h + (size_t)r * kH;
    len = kH; mode = 0;
  } else {
    const int r = rowid - kH - kD;
    src = cbf + (size_t)r * kH; dst = cb_h + (size_t)r * kH;
    len = kH; mode = 1;
  }
  float s = 0.0f;
  for (int i = tid; i < len; i += 128) { float v = src[i]; s += v * v; }
#pragma unroll
  for (int off = 16; off; off >>= 1) s += __shfl_xor(s, off, 32);
  if (lane == 0) red[wid] = s;
  __syncthreads();
  const float tot = red[0] + red[1] + red[2] + red[3];
  const float scale = (mode == 0)
      ? (1.0f / (sqrtf(tot) + kEps)) * rsqrtf((float)len)   // mp_linear weight
      : rsqrtf(tot / (float)len + kEps);                    // rms_norm codebook
  for (int i = tid; i < len; i += 128) dst[i] = (bf16_t)(src[i] * scale);
}

// GEMM1 + fused rmsnorm: h[m][n] = rmsnorm( sum_k z[b][k][l] * win_hat[n][k] )
// per WG: 128 tokens x 256 (full H); 8 waves, each wave: 16 M-rows x 16 N-tiles
__global__ __launch_bounds__(256) void k_gemm_h(const float* __restrict__ z,
                                                const bf16_t* __restrict__ win,
                                                bf16_t* __restrict__ hout) {
  __shared__ bf16_t As[128 * 40];   // [m][k], stride 40 (80B, 16B-aligned rows)
  __shared__ bf16_t Bs[256 * 40];   // [n][k]
  const int tid = threadIdx.x, lane = tid & 31, wid = tid >> 5;
  const int m0 = blockIdx.x * 128;
  const int b  = m0 / kL;
  const int l0 = m0 % kL;           // 128 | 4096 -> tile never crosses batch
  const float* zb = z + (size_t)b * kD * kL + l0;

  v8f acc[16];
#pragma unroll
  for (int t = 0; t < 16; ++t) acc[t] = vzero8();

  const int row  = lane & 15;
  const int kb   = (lane >> 4) * 16;
  const int ksel = (lane >> 4) * 16;

  for (int k0 = 0; k0 < kD; k0 += 32) {
    // B: win_hat rows are contiguous in K -> async copy to LDS
#pragma unroll
    for (int j = 0; j < 4; ++j) {
      const int c = j * 256 + tid;
      const int n = c >> 2, ko = (c & 3) * 8;
      lds_copy16(&Bs[n * 40 + ko], &win[(size_t)n * kD + k0 + ko]);
    }
    // A: z is [D][L] per batch -> coalesced rows, f32->bf16 convert in regs
#pragma unroll
    for (int i = 0; i < 16; ++i) {
      const int idx = i * 256 + tid;
      const int kk = idx >> 7, mm = idx & 127;
      As[mm * 40 + kk] = (bf16_t)zb[(size_t)(k0 + kk) * kL + mm];
    }
    lds_copy_wait<0>();
    __syncthreads();
    const v16bf a = ld_split(&As[(wid * 16 + row) * 40 + kb]);
    v16bf bcur = ld_cont(&Bs[(0 * 16 + row) * 40 + ksel]);
#pragma unroll
    for (int t = 0; t < 16; ++t) {
      v16bf bnxt = bcur;
      if (t < 15) bnxt = ld_cont(&Bs[((t + 1) * 16 + row) * 40 + ksel]);
      acc[t] = wmma_bf16(a, bcur, acc[t]);
      bcur = bnxt;
    }
    __syncthreads();
  }

  // rms_norm over H: row r (lanes<16) / r+8 (lanes>=16) spread over 16 lanes
  float rs[8];
#pragma unroll
  for (int r = 0; r < 8; ++r) {
    float s = 0.0f;
#pragma unroll
    for (int t = 0; t < 16; ++t) { const float v = acc[t][r]; s += v * v; }
#pragma unroll
    for (int off = 1; off < 16; off <<= 1) s += __shfl_xor(s, off, 32);
    rs[r] = rsqrtf(s * (1.0f / kH) + kEps);
  }
  const int mbase = m0 + wid * 16 + ((lane >> 4) ? 8 : 0);
  const int ncol  = lane & 15;
#pragma unroll
  for (int t = 0; t < 16; ++t) {
#pragma unroll
    for (int r = 0; r < 8; ++r) {
      hout[(size_t)(mbase + r) * kH + t * 16 + ncol] =
          (bf16_t)(acc[t][r] * rs[r]);
    }
  }
}

// GEMM2 + argmax: sim = h @ cb^T, running argmax over N, never materialized.
// per WG: 128 tokens; N chunks of 128, K chunks of 32 with double-buffered
// async staging: chunk kc+1 streams into LDS while WMMAs consume chunk kc.
__global__ __launch_bounds__(256) void k_argmax(const bf16_t* __restrict__ h,
                                                const bf16_t* __restrict__ cb,
                                                int* __restrict__ inds) {
  __shared__ bf16_t As[2][128 * 40];
  __shared__ bf16_t Bs[2][128 * 40];
  const int tid = threadIdx.x, lane = tid & 31, wid = tid >> 5;
  const int m0 = blockIdx.x * 128;
  const int row = lane & 15, kb = (lane >> 4) * 16, ksel = (lane >> 4) * 16;

  float best[8]; int bidx[8];
#pragma unroll
  for (int r = 0; r < 8; ++r) { best[r] = -3.4e38f; bidx[r] = 0; }

  for (int n0 = 0; n0 < kN; n0 += 128) {
    if (n0 + 128 < kN) __builtin_prefetch(&cb[(size_t)(n0 + 128) * kH], 0, 1);

    // stage(buf, k0): 4 async 16B copies per thread (2 A-chunks, 2 B-chunks)
    auto stage = [&](int buf, int k0) {
#pragma unroll
      for (int j = 0; j < 2; ++j) {
        const int c = j * 256 + tid;
        const int mm = c >> 2, ko = (c & 3) * 8;
        lds_copy16(&As[buf][mm * 40 + ko],
                   &h[(size_t)(m0 + mm) * kH + k0 + ko]);
      }
#pragma unroll
      for (int j = 0; j < 2; ++j) {
        const int c = j * 256 + tid;
        const int nn = c >> 2, ko = (c & 3) * 8;
        lds_copy16(&Bs[buf][nn * 40 + ko],
                   &cb[(size_t)(n0 + nn) * kH + k0 + ko]);
      }
    };

    v8f acc[8];
#pragma unroll
    for (int t = 0; t < 8; ++t) acc[t] = vzero8();

    stage(0, 0);
#pragma unroll
    for (int kc = 0; kc < 8; ++kc) {
      const int cur = kc & 1;
      if (kc < 7) {
        stage(cur ^ 1, (kc + 1) * 32);  // prefetch next chunk (async)
        lds_copy_wait<4>();             // oldest 4 copies (cur chunk) done
      } else {
        lds_copy_wait<0>();
      }
      __syncthreads();
      const v16bf a = ld_split(&As[cur][(wid * 16 + row) * 40 + kb]);
      v16bf bcur = ld_cont(&Bs[cur][(0 * 16 + row) * 40 + ksel]);
#pragma unroll
      for (int t = 0; t < 8; ++t) {
        v16bf bnxt = bcur;
        if (t < 7) bnxt = ld_cont(&Bs[cur][((t + 1) * 16 + row) * 40 + ksel]);
        acc[t] = wmma_bf16(a, bcur, acc[t]);
        bcur = bnxt;
      }
      __syncthreads();
    }
#pragma unroll
    for (int t = 0; t < 8; ++t) {
      const int n = n0 + t * 16 + (lane & 15);
#pragma unroll
      for (int r = 0; r < 8; ++r) {
        const float v = acc[t][r];
        if (v > best[r]) { best[r] = v; bidx[r] = n; }
      }
    }
  }
  // cross-lane argmax within each 16-lane half (ties -> smaller index)
#pragma unroll
  for (int r = 0; r < 8; ++r) {
    float bv = best[r]; int bi = bidx[r];
#pragma unroll
    for (int off = 1; off < 16; off <<= 1) {
      const float ov = __shfl_xor(bv, off, 32);
      const int   oi = __shfl_xor(bi, off, 32);
      if (ov > bv || (ov == bv && oi < bi)) { bv = ov; bi = oi; }
    }
    if ((lane & 15) == 0) {
      const int m = m0 + wid * 16 + r + ((lane >> 4) ? 8 : 0);
      inds[m] = bi;
    }
  }
}

// loss: sum over (m, h) of (cb_hat[ind[m]][h] - h_hat[m][h])^2
__global__ __launch_bounds__(256) void k_loss(const bf16_t* __restrict__ h,
                                              const bf16_t* __restrict__ cb,
                                              const int* __restrict__ inds,
                                              float* __restrict__ accum) {
  __shared__ float red[8];
  const int tid = threadIdx.x, lane = tid & 31, wid = tid >> 5;
  const int m = blockIdx.x * 8 + wid;            // one token per wave
  const bf16_t* hp = h + (size_t)m * kH;
  const bf16_t* qp = cb + (size_t)inds[m] * kH;
  const v8bf hv = *(const v8bf*)(hp + lane * 8);
  const v8bf qv = *(const v8bf*)(qp + lane * 8);
  float s = 0.0f;
#pragma unroll
  for (int j = 0; j < 8; ++j) {
    const float d = (float)qv[j] - (float)hv[j];
    s += d * d;
  }
#pragma unroll
  for (int off = 16; off; off >>= 1) s += __shfl_xor(s, off, 32);
  if (lane == 0) red[wid] = s;
  __syncthreads();
  if (tid == 0) {
    float t = 0.0f;
#pragma unroll
    for (int i = 0; i < 8; ++i) t += red[i];
    atomicAdd(accum, t);
  }
}

// GEMM3: out[b][d][l] = sum_k cb_hat[ind[m]][k] * wout_hat[d][k]
// per WG: 64 tokens x full D=512; waves 4(M) x 2(N halves of 256)
__global__ __launch_bounds__(256) void k_gemm_out(const bf16_t* __restrict__ cb,
                                                  const bf16_t* __restrict__ wout,
                                                  const int* __restrict__ inds,
                                                  float* __restrict__ out) {
  __shared__ bf16_t As[64 * 40];
  __shared__ bf16_t Bs[512 * 40];
  __shared__ int ind_l[64];
  const int tid = threadIdx.x, lane = tid & 31, wid = tid >> 5;
  const int wm = wid & 3, wn = wid >> 2;
  const int m0 = blockIdx.x * 64;
  if (tid < 64) ind_l[tid] = inds[m0 + tid];
  __syncthreads();

  const int row = lane & 15, kb = (lane >> 4) * 16, ksel = (lane >> 4) * 16;
  v8f acc[16];
#pragma unroll
  for (int t = 0; t < 16; ++t) acc[t] = vzero8();

  for (int k0 = 0; k0 < kH; k0 += 32) {
    {  // A: gather z_q rows from codebook (async, per-lane gathered addresses)
      const int mm = tid >> 2, ko = (tid & 3) * 8;
      lds_copy16(&As[mm * 40 + ko], &cb[(size_t)ind_l[mm] * kH + k0 + ko]);
    }
#pragma unroll
    for (int j = 0; j < 8; ++j) {
      const int c = j * 256 + tid;
      const int nn = c >> 2, ko = (c & 3) * 8;
      lds_copy16(&Bs[nn * 40 + ko], &wout[(size_t)nn * kH + k0 + ko]);
    }
    lds_copy_wait<0>();
    __syncthreads();
    const v16bf a = ld_split(&As[(wm * 16 + row) * 40 + kb]);
    v16bf bcur = ld_cont(&Bs[(wn * 256 + 0 * 16 + row) * 40 + ksel]);
#pragma unroll
    for (int t = 0; t < 16; ++t) {
      v16bf bnxt = bcur;
      if (t < 15)
        bnxt = ld_cont(&Bs[(wn * 256 + (t + 1) * 16 + row) * 40 + ksel]);
      acc[t] = wmma_bf16(a, bcur, acc[t]);
      bcur = bnxt;
    }
    __syncthreads();
  }

  const int mrow = m0 + wm * 16 + ((lane >> 4) ? 8 : 0);
#pragma unroll
  for (int t = 0; t < 16; ++t) {
    const int d = wn * 256 + t * 16 + (lane & 15);
#pragma unroll
    for (int r = 0; r < 8; ++r) {
      const int m = mrow + r;
      const int b = m >> 12;       // / 4096
      const int l = m & 4095;
      out[(size_t)b * kD * kL + (size_t)d * kL + l] = acc[t][r];
    }
  }
}

__global__ void k_fin(const int* __restrict__ inds,
                      const float* __restrict__ loss,
                      float* __restrict__ dout) {
  const int i = blockIdx.x * blockDim.x + threadIdx.x;
  if (i < kM) dout[OUT_IND + i] = (float)inds[i];
  if (i == 0)
    dout[OUT_LOSS] = (*loss) * (1.25f / ((float)kM * (float)kH));
}

// ------------------------------ launcher -----------------------------------

extern "C" void kernel_launch(void* const* d_in, const int* in_sizes, int n_in,
                              void* d_out, int out_size, void* d_ws,
                              size_t ws_size, hipStream_t stream) {
  const float* z     = (const float*)d_in[0];   // [B, D, L]
  const float* w_in  = (const float*)d_in[1];   // [H, D]
  const float* cbf   = (const float*)d_in[2];   // [N, H]
  const float* w_out = (const float*)d_in[3];   // [D, H]
  float* out = (float*)d_out;
  char* ws = (char*)d_ws;
  bf16_t* win_h  = (bf16_t*)(ws + WS_WIN);
  bf16_t* wout_h = (bf16_t*)(ws + WS_WOUT);
  bf16_t* cb_h   = (bf16_t*)(ws + WS_CB);
  bf16_t* h      = (bf16_t*)(ws + WS_H);
  int*    inds   = (int*)(ws + WS_IND);
  float*  loss   = (float*)(ws + WS_LOSS);

  k_zero<<<1, 32, 0, stream>>>(loss);
  k_prep<<<kH + kD + kN, 128, 0, stream>>>(w_in, cbf, w_out, win_h, wout_h, cb_h);
  k_gemm_h<<<kM / 128, 256, 0, stream>>>(z, win_h, h);
  k_argmax<<<kM / 128, 256, 0, stream>>>(h, cb_h, inds);
  k_loss<<<kM / 8, 256, 0, stream>>>(h, cb_h, inds, loss);
  k_gemm_out<<<kM / 64, 256, 0, stream>>>(cb_h, wout_h, inds, out);
  k_fin<<<(kM + 255) / 256, 256, 0, stream>>>(inds, loss, out);
}
